// LSTMModel_88596585382367
// MI455X (gfx1250) — matrix operand order repaired and checked by reading, not compiled, not run on hardware
//
#include <hip/hip_runtime.h>
#include <hip/hip_bf16.h>

typedef __attribute__((ext_vector_type(16))) __bf16 v16bf;
typedef __attribute__((ext_vector_type(8)))  __bf16 v8bf;
typedef __attribute__((ext_vector_type(8)))  float  v8f;

#define H_   50      // hidden size
#define IN_  3       // input features
#define OUT_ 7       // fc out
#define T_   512     // sequence length
#define BM   16      // batch rows per workgroup (one WMMA M-tile)
#define K0S  64      // layer0 A row stride: [h0(50) | x(3) | pad]
#define K1S  128     // layer1 A row stride: [h1(50) | y(50) | pad]
#define NT   13      // N tiles of 16 covering 200 gate columns
#define GS   208     // gate buffer row stride (13*16)
#define NTHREADS 256 // 8 wave32
#define EWTHREADS 200  // threads doing elementwise: 4 elements each (800 total)
#define XBASE 200      // x-staging handled by threads 200..247 (idle waves)

__device__ __forceinline__ float frcp(float x) { return __builtin_amdgcn_rcpf(x); }
__device__ __forceinline__ float fsigmoid(float x) {
  return frcp(1.0f + __expf(-x));                 // v_exp + v_add + v_rcp
}
__device__ __forceinline__ float ftanh(float x) {
  float e = __expf(2.0f * x);
  return (e - 1.0f) * frcp(e + 1.0f);             // v_exp + 2 add + rcp + mul
}

// Load one 16x32 bf16 A-fragment (ISA 16-bit A layout) from LDS row-major buf.
// lanes 0-15: row M=lane, elements = K {kb..kb+7, kb+16..kb+23}
// lanes 16-31: row M=lane-16, elements = K {kb+8..kb+15, kb+24..kb+31}
__device__ __forceinline__ v16bf load_a_frag(const __bf16* buf, int stride, int kbase) {
  int lane = threadIdx.x & 31;
  int row  = lane & 15;
  int off  = row * stride + kbase + ((lane & 16) ? 8 : 0);
  v8bf lo = *(const v8bf*)(buf + off);
  v8bf hi = *(const v8bf*)(buf + off + 16);
  return __builtin_shufflevector(lo, hi, 0,1,2,3,4,5,6,7,8,9,10,11,12,13,14,15);
}

// B fragment (32x16 bf16, ISA layout): lanes 0-15 -> col N=n hold K {0..7,16..23} of
// the chunk, lanes 16-31 hold K {8..15,24..31}.  Layer0 B = [Whh0^T ; Wih0^T ; 0].
__device__ __forceinline__ v16bf make_b0_frag(const float* __restrict__ Whh0,
                                              const float* __restrict__ Wih0,
                                              int tile, int chunk) {
  int lane = threadIdx.x & 31;
  int n  = tile * 16 + (lane & 15);
  int kb = chunk * 32 + ((lane & 16) ? 8 : 0);
  v16bf f;
  #pragma unroll
  for (int i = 0; i < 16; ++i) {
    int k = kb + (i < 8 ? i : i + 8);
    float v = 0.0f;
    if (n < 4 * H_) {
      if (k < H_)            v = Whh0[n * H_ + k];
      else if (k < H_ + IN_) v = Wih0[n * IN_ + (k - H_)];
    }
    f[i] = (__bf16)v;
  }
  return f;
}

// Layer1 B = [Whh1^T (K 0..49) ; Wih1^T (K 50..99) ; 0]
__device__ __forceinline__ v16bf make_b1_frag(const float* __restrict__ Whh1,
                                              const float* __restrict__ Wih1,
                                              int tile, int chunk) {
  int lane = threadIdx.x & 31;
  int n  = tile * 16 + (lane & 15);
  int kb = chunk * 32 + ((lane & 16) ? 8 : 0);
  v16bf f;
  #pragma unroll
  for (int i = 0; i < 16; ++i) {
    int k = kb + (i < 8 ? i : i + 8);
    float v = 0.0f;
    if (n < 4 * H_) {
      if (k < H_)           v = Whh1[n * H_ + k];
      else if (k < 2 * H_)  v = Wih1[n * H_ + (k - H_)];
    }
    f[i] = (__bf16)v;
  }
  return f;
}

__global__ __launch_bounds__(NTHREADS)
void lstm2_wmma_kernel(const float* __restrict__ x,
                       const float* __restrict__ Wih0, const float* __restrict__ Whh0,
                       const float* __restrict__ bih0, const float* __restrict__ bhh0,
                       const float* __restrict__ Wih1, const float* __restrict__ Whh1,
                       const float* __restrict__ bih1, const float* __restrict__ bhh1,
                       const float* __restrict__ Wfc,  const float* __restrict__ bfc,
                       float* __restrict__ out)
{
  __shared__ __bf16 a0[BM * K0S];   // layer0 A: [h0 | x_t | 0]
  __shared__ __bf16 a1[BM * K1S];   // layer1 A: [h1 | y   | 0]
  __shared__ float  gb[BM * GS];    // gate pre-activations (f32)
  __shared__ float  c0[BM * H_];    // cell states
  __shared__ float  c1[BM * H_];

  const int tid   = threadIdx.x;
  // scalar wave id -> tile guards become s_cmp/s_cbranch, EXEC stays all-ones
  const int wave  = __builtin_amdgcn_readfirstlane(tid >> 5);
  const int lane  = tid & 31;
  const int b0row = blockIdx.x * BM;

  // zero LDS state (pads stay zero forever)
  for (int i = tid; i < BM * K0S; i += NTHREADS) a0[i] = (__bf16)0.0f;
  for (int i = tid; i < BM * K1S; i += NTHREADS) a1[i] = (__bf16)0.0f;
  for (int i = tid; i < BM * H_;  i += NTHREADS) { c0[i] = 0.0f; c1[i] = 0.0f; }

  // ---- per-thread elementwise offsets, computed once.
  // threads 0..199 own elements 4*tid .. 4*tid+3  (uniform count: one branch)
  const bool doEW = (tid < EWTHREADS);
  int oGB[4], oC[4], oA0[4], oA1y[4], oA1h[4];
  if (doEW) {
    #pragma unroll
    for (int i = 0; i < 4; ++i) {
      int idx = tid * 4 + i;
      int b = idx / H_, h = idx % H_;
      oGB[i]  = b * GS + h;
      oC[i]   = idx;
      oA0[i]  = b * K0S + h;
      oA1y[i] = b * K1S + H_ + h;
      oA1h[i] = b * K1S + h;
    }
  }

  // ---- x staging on threads 200..247 (concurrent with elementwise waves)
  const bool doX = (tid >= XBASE) && (tid < XBASE + BM * IN_);
  const float* xp = x;
  int xdst = 0;
  if (doX) {
    int b = (tid - XBASE) / IN_, j = (tid - XBASE) % IN_;
    xp   = x + (size_t)(b0row + b) * T_ * IN_ + j;
    xdst = b * K0S + H_ + j;
  }

  // ---- per-wave weight fragments: 2 N-tiles per wave, register-resident
  v16bf b0f[2][2];
  v16bf b1f[2][4];
  float bias0[2], bias1[2];
  #pragma unroll
  for (int s = 0; s < 2; ++s) {
    int tt = wave * 2 + s;
    bias0[s] = 0.0f; bias1[s] = 0.0f;
    if (tt < NT) {
      #pragma unroll
      for (int c = 0; c < 2; ++c) b0f[s][c] = make_b0_frag(Whh0, Wih0, tt, c);
      #pragma unroll
      for (int c = 0; c < 4; ++c) b1f[s][c] = make_b1_frag(Whh1, Wih1, tt, c);
      int n = tt * 16 + (lane & 15);
      if (n < 4 * H_) {
        bias0[s] = bih0[n] + bhh0[n];
        bias1[s] = bih1[n] + bhh1[n];
      }
    }
  }

  // pre-stage x_0
  if (doX) a0[xdst] = (__bf16)xp[0];
  __syncthreads();

  for (int t = 0; t < T_; ++t) {
    // ---------- layer 0: gates = [h0|x_t] @ B0 + b0 ----------
    {
      v16bf A0 = load_a_frag(a0, K0S, 0);
      v16bf A1 = load_a_frag(a0, K0S, 32);
      #pragma unroll
      for (int s = 0; s < 2; ++s) {
        int tt = wave * 2 + s;
        if (tt < NT) {                   // scalar branch: EXEC all-ones inside
          v8f acc;
          #pragma unroll
          for (int r = 0; r < 8; ++r) acc[r] = bias0[s];
          acc = __builtin_amdgcn_wmma_f32_16x16x32_bf16(false, A0, false, b0f[s][0], (short)0, acc, false, false);
          acc = __builtin_amdgcn_wmma_f32_16x16x32_bf16(false, A1, false, b0f[s][1], (short)0, acc, false, false);
          int n  = tt * 16 + (lane & 15);
          int mb = (lane & 16) ? 8 : 0;  // C/D layout: M = r + 8*(lane>=16)
          #pragma unroll
          for (int r = 0; r < 8; ++r) gb[(mb + r) * GS + n] = acc[r];
        }
      }
    }
    __syncthreads();

    // layer0 elementwise: c0/h0 update; h0 -> a0[:,0..49] and a1[:,50..99]
    if (doEW) {
      #pragma unroll
      for (int i = 0; i < 4; ++i) {
        const float* g = gb + oGB[i];
        float gi = fsigmoid(g[0]);
        float gf = fsigmoid(g[H_]);
        float gg = ftanh(g[2 * H_]);
        float go = fsigmoid(g[3 * H_]);
        float c  = gf * c0[oC[i]] + gi * gg;
        c0[oC[i]] = c;
        float hv = go * ftanh(c);
        __bf16 hb = (__bf16)hv;
        a0[oA0[i]]  = hb;   // recurrent input, layer0
        a1[oA1y[i]] = hb;   // feed-forward input, layer1
      }
    }
    __syncthreads();

    // ---------- layer 1: gates = [h1|y] @ B1 + b1 ----------
    {
      v16bf A0 = load_a_frag(a1, K1S, 0);
      v16bf A1 = load_a_frag(a1, K1S, 32);
      v16bf A2 = load_a_frag(a1, K1S, 64);
      v16bf A3 = load_a_frag(a1, K1S, 96);
      #pragma unroll
      for (int s = 0; s < 2; ++s) {
        int tt = wave * 2 + s;
        if (tt < NT) {
          v8f acc;
          #pragma unroll
          for (int r = 0; r < 8; ++r) acc[r] = bias1[s];
          acc = __builtin_amdgcn_wmma_f32_16x16x32_bf16(false, A0, false, b1f[s][0], (short)0, acc, false, false);
          acc = __builtin_amdgcn_wmma_f32_16x16x32_bf16(false, A1, false, b1f[s][1], (short)0, acc, false, false);
          acc = __builtin_amdgcn_wmma_f32_16x16x32_bf16(false, A2, false, b1f[s][2], (short)0, acc, false, false);
          acc = __builtin_amdgcn_wmma_f32_16x16x32_bf16(false, A3, false, b1f[s][3], (short)0, acc, false, false);
          int n  = tt * 16 + (lane & 15);
          int mb = (lane & 16) ? 8 : 0;
          #pragma unroll
          for (int r = 0; r < 8; ++r) gb[(mb + r) * GS + n] = acc[r];
        }
      }
    }
    __syncthreads();

    // layer1 elementwise (threads 0..199) + stage x_{t+1} (threads 200..247)
    {
      const bool last = (t == T_ - 1);
      if (doEW) {
        #pragma unroll
        for (int i = 0; i < 4; ++i) {
          float* g = gb + oGB[i];
          float gi = fsigmoid(g[0]);
          float gf = fsigmoid(g[H_]);
          float gg = ftanh(g[2 * H_]);
          float go = fsigmoid(g[3 * H_]);
          float c  = gf * c1[oC[i]] + gi * gg;
          c1[oC[i]] = c;
          float hv = go * ftanh(c);
          a1[oA1h[i]] = (__bf16)hv;
          if (last) g[0] = hv;   // stash f32 h_last (only this thread touches it)
        }
      }
      if (doX && t + 1 < T_) {
        a0[xdst] = (__bf16)xp[(t + 1) * IN_];
        if (t + 2 < T_) __builtin_prefetch(xp + (t + 2) * IN_, 0, 0);
      }
    }
    __syncthreads();
  }

  // final FC on h_last (f32 in gb[:,0..49]): out[b][o] = h . Wfc[o] + bfc[o]
  if (tid < BM * OUT_) {
    int b = tid / OUT_, o = tid % OUT_;
    float s = bfc[o];
    const float* hrow = gb + b * GS;
    #pragma unroll
    for (int h = 0; h < H_; ++h) s += hrow[h] * Wfc[o * H_ + h];
    out[(size_t)(b0row + b) * OUT_ + o] = s;
  }
}

extern "C" void kernel_launch(void* const* d_in, const int* in_sizes, int n_in,
                              void* d_out, int out_size, void* d_ws, size_t ws_size,
                              hipStream_t stream) {
  (void)n_in; (void)out_size; (void)d_ws; (void)ws_size;
  const float* x    = (const float*)d_in[0];
  const float* Wih0 = (const float*)d_in[1];
  const float* Whh0 = (const float*)d_in[2];
  const float* bih0 = (const float*)d_in[3];
  const float* bhh0 = (const float*)d_in[4];
  const float* Wih1 = (const float*)d_in[5];
  const float* Whh1 = (const float*)d_in[6];
  const float* bih1 = (const float*)d_in[7];
  const float* bhh1 = (const float*)d_in[8];
  const float* Wfc  = (const float*)d_in[9];
  const float* bfc  = (const float*)d_in[10];
  float* out = (float*)d_out;

  int B = in_sizes[0] / (T_ * IN_);   // 4096
  dim3 grid(B / BM), block(NTHREADS);
  hipLaunchKernelGGL(lstm2_wmma_kernel, grid, block, 0, stream,
                     x, Wih0, Whh0, bih0, bhh0, Wih1, Whh1, bih1, bhh1, Wfc, bfc, out);
}
